// Attention_54778012893378
// MI455X (gfx1250) — compile-verified, matrix operand order checked by reading, not run on hardware
//
#include <hip/hip_runtime.h>
#include <hip/hip_bf16.h>

typedef __bf16        v16bf __attribute__((ext_vector_type(16)));
typedef float         v8f   __attribute__((ext_vector_type(8)));
typedef unsigned int  u32x4 __attribute__((ext_vector_type(4)));

union Frag { v16bf v; u32x4 u[2]; };

__device__ __forceinline__ unsigned short f2bf(float f) {
  union { float f; unsigned u; } x; x.f = f;
  unsigned r = x.u + 0x7FFFu + ((x.u >> 16) & 1u);
  return (unsigned short)(r >> 16);
}
__device__ __forceinline__ float bf2f(unsigned short b) {
  union { unsigned u; float f; } x; x.u = ((unsigned)b) << 16;
  return x.f;
}
__device__ __forceinline__ void lds_frag(Frag& f, const unsigned short* base) {
  f.u[0] = *(const u32x4*)(base);
  f.u[1] = *(const u32x4*)(base + 16);
}
#define WMMA_BF16(a, b, c) \
  __builtin_amdgcn_wmma_f32_16x16x32_bf16(false, (a).v, false, (b).v, (short)0, (c), false, false)

// ---- CDNA5 async global->LDS copy (bypasses VGPRs, tracked by ASYNCcnt) ----
#define USE_ASYNC_LDS 1
__device__ __forceinline__ void async_copy_b128(const void* gptr, void* lptr) {
#if USE_ASYNC_LDS
  unsigned loff = (unsigned)(unsigned long long)lptr;  // LDS aperture: low 32 bits
  asm volatile("global_load_async_to_lds_b128 %0, %1, off"
               :: "v"(loff), "v"((unsigned long long)gptr) : "memory");
#else
  *(u32x4*)lptr = *(const u32x4*)gptr;
#endif
}
__device__ __forceinline__ void async_wait() {
#if USE_ASYNC_LDS
  asm volatile("s_wait_asynccnt 0x0" ::: "memory");
#endif
}

// ---------------- fp32 -> bf16 conversion ----------------
__global__ void f32_to_bf16(const float* __restrict__ in,
                            unsigned short* __restrict__ out, int n) {
  for (int i = blockIdx.x * blockDim.x + threadIdx.x; i < n;
       i += gridDim.x * blockDim.x)
    out[i] = f2bf(in[i]);
}

// ---------------- bf16 GEMM via WMMA: C[M,N] = A[M,K] @ B[K,N] ----------------
// 128x128 block tile, 8 waves (4x2), each wave 32x64 = 2x4 WMMA frags.
template <bool OUT_F32>
__global__ __launch_bounds__(256)
void gemm_bf16_wmma(const unsigned short* __restrict__ A,
                    const unsigned short* __restrict__ B,
                    void* __restrict__ C, int M, int N, int K) {
  constexpr int BM = 128, BN = 128, BK = 32, LD = 40;
  __shared__ __align__(16) unsigned short As[BM * LD];      // [m][k]
  __shared__ __align__(16) unsigned short Bs[BN * LD];      // transposed: [n][k]
  const int tid  = threadIdx.x;
  const int lane = tid & 31, wave = tid >> 5;
  const int half = lane >> 4, r = lane & 15;
  const int wrow = wave >> 1, wcol = wave & 1;
  const int row0 = blockIdx.y * BM, col0 = blockIdx.x * BN;

  v8f acc[2][4];
#pragma unroll
  for (int i = 0; i < 2; ++i)
#pragma unroll
    for (int j = 0; j < 4; ++j)
#pragma unroll
      for (int e = 0; e < 8; ++e) acc[i][j][e] = 0.f;

  for (int kb = 0; kb < K; kb += BK) {
    __syncthreads();
    // stage A: 128x32 bf16 via async global->LDS (512 chunks of 8, 2/thread)
#pragma unroll
    for (int i = 0; i < 2; ++i) {
      int lin = tid + i * 256;
      int ar = lin >> 2, ac = (lin & 3) * 8;
      async_copy_b128(&A[(size_t)(row0 + ar) * K + kb + ac],
                      &As[ar * LD + ac]);
    }
    // stage B transposed: 32x128 -> Bs[n][k] (needs VGPR round-trip)
#pragma unroll
    for (int i = 0; i < 2; ++i) {
      int lin = tid + i * 256;
      int br = lin >> 4, bc = (lin & 15) * 8;
      u32x4 pk = *(const u32x4*)&B[(size_t)(kb + br) * N + col0 + bc];
      const unsigned short* pv = (const unsigned short*)&pk;
#pragma unroll
      for (int j = 0; j < 8; ++j) Bs[(bc + j) * LD + br] = pv[j];
    }
    // prefetch next K-panel into cache (global_prefetch_b8)
    if (kb + BK < K) {
      __builtin_prefetch(&A[(size_t)(row0 + (tid >> 1)) * K + kb + BK], 0, 3);
      __builtin_prefetch(&B[(size_t)(kb + BK + (tid >> 3)) * N + col0], 0, 3);
    }
    async_wait();
    __syncthreads();

    Frag a[2], b[4];
#pragma unroll
    for (int i = 0; i < 2; ++i)
      lds_frag(a[i], &As[(wrow * 32 + i * 16 + r) * LD + half * 8]);
#pragma unroll
    for (int j = 0; j < 4; ++j)
      lds_frag(b[j], &Bs[(wcol * 64 + j * 16 + r) * LD + half * 8]);
#pragma unroll
    for (int i = 0; i < 2; ++i)
#pragma unroll
      for (int j = 0; j < 4; ++j)
        acc[i][j] = WMMA_BF16(a[i], b[j], acc[i][j]);
  }

#pragma unroll
  for (int i = 0; i < 2; ++i)
#pragma unroll
    for (int j = 0; j < 4; ++j)
#pragma unroll
      for (int e = 0; e < 8; ++e) {
        int m = row0 + wrow * 32 + i * 16 + e + 8 * half;
        int n = col0 + wcol * 64 + j * 16 + r;
        if (OUT_F32)
          ((float*)C)[(size_t)m * N + n] = acc[i][j][e];
        else
          ((unsigned short*)C)[(size_t)m * N + n] = f2bf(acc[i][j][e]);
      }
}

// ---------------- RoPE on Q and K (in workspace, bf16) ----------------
__global__ void rope_kernel(unsigned short* __restrict__ q,
                            unsigned short* __restrict__ k,
                            const float* __restrict__ fc,
                            const float* __restrict__ fs) {
  int idx = blockIdx.x * blockDim.x + threadIdx.x;  // S*H*64
  int i = idx & 63;
  int h = (idx >> 6) & 15;
  int s = idx >> 10;
  float c = fc[s * 64 + i], sn = fs[s * 64 + i];
  size_t base = (size_t)s * 2048 + h * 128 + 2 * i;
  float te = bf2f(q[base]), to = bf2f(q[base + 1]);
  q[base]     = f2bf(te * c - to * sn);
  q[base + 1] = f2bf(te * sn + to * c);
  te = bf2f(k[base]); to = bf2f(k[base + 1]);
  k[base]     = f2bf(te * c - to * sn);
  k[base + 1] = f2bf(te * sn + to * c);
}

// ---------------- causal flash attention, WMMA for QK^T and P@V ----------------
__global__ __launch_bounds__(256)
void flash_attn(const unsigned short* __restrict__ Q,
                const unsigned short* __restrict__ Kb,
                const unsigned short* __restrict__ Vb,
                const float* __restrict__ seq_scale,
                unsigned short* __restrict__ O) {
  constexpr int LDK = 136, LDV = 40, LDP = 40;
  __shared__ __align__(16) unsigned short Kt[32 * LDK];       // [key][hd]
  __shared__ __align__(16) unsigned short Vt[128 * LDV];      // transposed: [hd][key]
  __shared__ __align__(16) unsigned short Pl[8 * 16 * LDP];   // per-wave P patch
  const int tid = threadIdx.x, lane = tid & 31, wave = tid >> 5;
  const int half = lane >> 4, r = lane & 15;
  const int h = blockIdx.y, qt = blockIdx.x;
  const int q0 = qt * 128 + wave * 16;   // 16 query rows per wave
  const size_t hoff = (size_t)h * 128;

  // Q frags straight from global: A-frag layout == row-major 16B chunks
  Frag aq[4];
  {
    const unsigned short* qp = &Q[(size_t)(q0 + r) * 2048 + hoff];
#pragma unroll
    for (int c = 0; c < 4; ++c) {
      aq[c].u[0] = *(const u32x4*)(qp + c * 32 + half * 8);
      aq[c].u[1] = *(const u32x4*)(qp + c * 32 + half * 8 + 16);
    }
  }
  const float hs = seq_scale[h] * 0.0883883476483184f;  // 1/sqrt(128)
  float rs[8];
#pragma unroll
  for (int e = 0; e < 8; ++e)
    rs[e] = __logf((float)(q0 + e + 8 * half) + 2.0f) * hs;

  float mrow[8], lrow[8];
  v8f o[8];
#pragma unroll
  for (int e = 0; e < 8; ++e) { mrow[e] = -3.0e38f; lrow[e] = 0.f; }
#pragma unroll
  for (int f = 0; f < 8; ++f)
#pragma unroll
    for (int e = 0; e < 8; ++e) o[f][e] = 0.f;

  const int ntiles = (qt + 1) * 4;  // causal: keys up to (qt+1)*128
  for (int t = 0; t < ntiles; ++t) {
    const int kv0 = t * 32;
    __syncthreads();
    // stage K tile 32x128 via async global->LDS (row-major == Bt for QK^T)
#pragma unroll
    for (int i = 0; i < 2; ++i) {
      int lin = tid + i * 256;
      int kr = lin >> 4, kc = (lin & 15) * 8;
      async_copy_b128(&Kb[(size_t)(kv0 + kr) * 2048 + hoff + kc],
                      &Kt[kr * LDK + kc]);
    }
    // stage V transposed: Vt[hd][key] (needs VGPR round-trip)
#pragma unroll
    for (int i = 0; i < 2; ++i) {
      int lin = tid + i * 256;
      int vr = lin >> 4, vc = (lin & 15) * 8;
      u32x4 pk = *(const u32x4*)&Vb[(size_t)(kv0 + vr) * 2048 + hoff + vc];
      const unsigned short* pv = (const unsigned short*)&pk;
#pragma unroll
      for (int j = 0; j < 8; ++j) Vt[(vc + j) * LDV + vr] = pv[j];
    }
    async_wait();
    __syncthreads();

    // scores: S[16 q][32 keys] = Q @ K^T, accumulate over 4 hd-chunks
    v8f s0, s1;
#pragma unroll
    for (int e = 0; e < 8; ++e) { s0[e] = 0.f; s1[e] = 0.f; }
#pragma unroll
    for (int c = 0; c < 4; ++c) {
      Frag bk0, bk1;
      lds_frag(bk0, &Kt[r * LDK + c * 32 + half * 8]);
      lds_frag(bk1, &Kt[(16 + r) * LDK + c * 32 + half * 8]);
      s0 = WMMA_BF16(aq[c], bk0, s0);
      s1 = WMMA_BF16(aq[c], bk1, s1);
    }

    // online softmax (rows live in 16-lane halves; shfl_xor 1/2/4/8 stays in-half)
    const int key0 = kv0 + r, key1 = kv0 + 16 + r;
    float p0[8], p1[8], alpha[8];
#pragma unroll
    for (int e = 0; e < 8; ++e) {
      int qrow = q0 + e + 8 * half;
      float v0 = (key0 <= qrow) ? s0[e] * rs[e] : -1.0e9f;
      float v1 = (key1 <= qrow) ? s1[e] * rs[e] : -1.0e9f;
      float mx = fmaxf(v0, v1);
      mx = fmaxf(mx, __shfl_xor(mx, 1, 32));
      mx = fmaxf(mx, __shfl_xor(mx, 2, 32));
      mx = fmaxf(mx, __shfl_xor(mx, 4, 32));
      mx = fmaxf(mx, __shfl_xor(mx, 8, 32));
      float mnew = fmaxf(mrow[e], mx);
      alpha[e] = __expf(mrow[e] - mnew);
      p0[e] = __expf(v0 - mnew);
      p1[e] = __expf(v1 - mnew);
      float sum = p0[e] + p1[e];
      sum += __shfl_xor(sum, 1, 32);
      sum += __shfl_xor(sum, 2, 32);
      sum += __shfl_xor(sum, 4, 32);
      sum += __shfl_xor(sum, 8, 32);
      lrow[e] = lrow[e] * alpha[e] + sum;
      mrow[e] = mnew;
    }
#pragma unroll
    for (int f = 0; f < 8; ++f)
#pragma unroll
      for (int e = 0; e < 8; ++e) o[f][e] *= alpha[e];

    // reshape P: D-frag (lane=col) -> A-frag (lane=row) via wave-private LDS
    unsigned short* Pw = &Pl[wave * 16 * LDP];
#pragma unroll
    for (int e = 0; e < 8; ++e) {
      int row = e + 8 * half;
      Pw[row * LDP + r]      = f2bf(p0[e]);
      Pw[row * LDP + 16 + r] = f2bf(p1[e]);
    }
    Frag ap;
    lds_frag(ap, &Pw[r * LDP + half * 8]);
    // O[16 q][128 hd] += P[16 q][32 k] @ V[32 k][128 hd]
#pragma unroll
    for (int f = 0; f < 8; ++f) {
      Frag bv;
      lds_frag(bv, &Vt[(f * 16 + r) * LDV + half * 8]);
      o[f] = WMMA_BF16(ap, bv, o[f]);
    }
  }

  // normalize + store bf16
#pragma unroll
  for (int e = 0; e < 8; ++e) {
    int qrow = q0 + e + 8 * half;
    float inv = 1.0f / fmaxf(lrow[e], 1e-20f);
#pragma unroll
    for (int f = 0; f < 8; ++f)
      O[(size_t)qrow * 2048 + hoff + f * 16 + r] = f2bf(o[f][e] * inv);
  }
}

// ---------------- driver ----------------
extern "C" void kernel_launch(void* const* d_in, const int* in_sizes, int n_in,
                              void* d_out, int out_size, void* d_ws, size_t ws_size,
                              hipStream_t stream) {
  const float* x  = (const float*)d_in[0];
  const float* wq = (const float*)d_in[1];
  const float* wk = (const float*)d_in[2];
  const float* wv = (const float*)d_in[3];
  const float* wo = (const float*)d_in[4];
  const float* seq_scale = (const float*)d_in[5];
  const float* fc = (const float*)d_in[6];
  const float* fs = (const float*)d_in[7];

  const int S = 2048, D = 2048;
  const int n = S * D;
  const size_t CH = (size_t)n * sizeof(unsigned short);  // 8 MiB per bf16 buffer
  char* ws = (char*)d_ws;
  unsigned short* xb  = (unsigned short*)(ws + 0 * CH);
  unsigned short* wqb = (unsigned short*)(ws + 1 * CH);
  unsigned short* wkb = (unsigned short*)(ws + 2 * CH);
  unsigned short* wvb = (unsigned short*)(ws + 3 * CH);
  unsigned short* wob = (unsigned short*)(ws + 4 * CH);
  unsigned short* qb  = (unsigned short*)(ws + 5 * CH);
  unsigned short* kb  = (unsigned short*)(ws + 6 * CH);
  unsigned short* vb  = (unsigned short*)(ws + 7 * CH);
  unsigned short* att = (unsigned short*)(ws + 8 * CH);

  f32_to_bf16<<<1024, 256, 0, stream>>>(x,  xb,  n);
  f32_to_bf16<<<1024, 256, 0, stream>>>(wq, wqb, n);
  f32_to_bf16<<<1024, 256, 0, stream>>>(wk, wkb, n);
  f32_to_bf16<<<1024, 256, 0, stream>>>(wv, wvb, n);
  f32_to_bf16<<<1024, 256, 0, stream>>>(wo, wob, n);

  dim3 gg(D / 128, S / 128);
  gemm_bf16_wmma<false><<<gg, 256, 0, stream>>>(xb, wqb, qb, S, D, D);
  gemm_bf16_wmma<false><<<gg, 256, 0, stream>>>(xb, wkb, kb, S, D, D);
  gemm_bf16_wmma<false><<<gg, 256, 0, stream>>>(xb, wvb, vb, S, D, D);

  rope_kernel<<<(S * 16 * 64) / 256, 256, 0, stream>>>(qb, kb, fc, fs);

  flash_attn<<<dim3(S / 128, 16), 256, 0, stream>>>(qb, kb, vb, seq_scale, att);

  gemm_bf16_wmma<true><<<gg, 256, 0, stream>>>(att, wob, d_out, S, D, D);
}
